// LSTM_CELL_1236950582232
// MI455X (gfx1250) — matrix operand order repaired and checked by reading, not compiled, not run on hardware
//
#include <hip/hip_runtime.h>
#include <math.h>

// ---------------------------------------------------------------------------
// LSTM scan for MI455X (gfx1250, wave32, WMMA + Tensor Data Mover).
// Phase 1: xg[S*B,4H] = x[S*B,D] @ W_ih^T  (bf16 WMMA, f32 accum, bf16 out)
// Phase 2: persistent 32-block kernel, per-step h@W_hh^T WMMA + fused gates.
//          W_hh slice resident in LDS; h and xg slices staged per step by the
//          TDM (tensor_load_to_lds, TENSORcnt); global spin barrier per step.
// Workspace: [ xg bf16 : 512MB ][ h ping-pong bf16 : 128KB ][ counter ]
// ---------------------------------------------------------------------------

#define S_LEN 2048
#define BATCH 64
#define DIM   512
#define HID   512
#define GATES (4 * HID)          // 2048
#define NBLK  32                 // persistent blocks (1 per WGP, ~156KB LDS)
#define HC    (HID / NBLK)       // 16 H-columns per block
#define NCOL  (4 * HC)           // 64 gate columns per block

typedef __bf16 bf16;
typedef __attribute__((ext_vector_type(16))) __bf16 v16bf;
typedef __attribute__((ext_vector_type(8)))  float  v8f;
typedef __attribute__((ext_vector_type(4)))  unsigned int u32x4;
typedef __attribute__((ext_vector_type(8)))  int  i32x8;
typedef __attribute__((ext_vector_type(4)))  int  i32x4;

// --- packed f32 -> 2 x bf16 (lowers to v_cvt_pk_bf16_f32) ------------------
static __device__ inline unsigned pack_bf16x2(float a, float b) {
  union { unsigned u; __bf16 h[2]; } p;
  p.h[0] = (__bf16)a;
  p.h[1] = (__bf16)b;
  return p.u;
}

// --- WMMA fragment loads from LDS (contiguous k order, same for A and B) -------
// A (16x32 bf16, row M = lane&15): lane holds K = k0+hi*16 + {0..15}
static __device__ inline v16bf lds_a_frag(const bf16* p) {
  union { v16bf v; uint4 q[2]; } u;
  u.q[0] = *(const uint4*)(p);
  u.q[1] = *(const uint4*)(p + 8);
  return u.v;
}
// B (32x16 bf16, col N = lane&15): lane holds K = k0+hi*16 + {0..15}
static __device__ inline v16bf lds_b_frag(const bf16* p) {
  union { v16bf v; uint4 q[2]; } u;
  u.q[0] = *(const uint4*)(p);
  u.q[1] = *(const uint4*)(p + 8);
  return u.v;
}

static __device__ inline v8f wmma_bf16(v16bf a, v16bf b, v8f c) {
  return __builtin_amdgcn_wmma_f32_16x16x32_bf16(
      false, a, false, b, (short)0, c, false, false);
}

static __device__ inline float sigmoidf_(float x) {
  return 1.0f / (1.0f + __expf(-x));
}

// ---------------------------------------------------------------------------
// Tensor Data Mover: load a (up to 3D) bf16 tile into LDS.
//   dim0: contiguous elements per row, dim1: rows (stride0 elems apart),
//   dim2: planes (stride1 elems apart, 0 => 2D). Tensor extents == tile
//   extents (no OOB clipping needed). D# layout per CDNA5 ISA.
// This toolchain exposes the 6-arg builtin:
//   (u32x4 g0, i32x8 g1, i32x4 g2, i32x4 g3, i32x8 pad, i32 cpol)
// ---------------------------------------------------------------------------
static __device__ inline void tdm_load_bf16(unsigned lds_addr,
                                            unsigned long long gaddr,
                                            unsigned dim0, unsigned dim1,
                                            unsigned dim2,
                                            unsigned long long stride0,
                                            unsigned long long stride1) {
  u32x4 g0;
  g0[0] = 1u;                                   // count=1, user descriptor
  g0[1] = lds_addr;                             // LDS byte address
  g0[2] = (unsigned)gaddr;                      // global_addr[31:0]
  g0[3] = (unsigned)((gaddr >> 32) & 0x1FFFFFFull) | 0x80000000u; // type=2
  i32x8 g1;
  g1[0] = 0x10000;                              // data_size=1 (2 bytes)
  g1[1] = (int)(dim0 << 16);                    // tensor_dim0[15:0]
  g1[2] = (int)((dim0 >> 16) | (dim1 << 16));   // tdim0[31:16] | tdim1[15:0]
  g1[3] = (int)((dim1 >> 16) | (dim0 << 16));   // tdim1[31:16] | tile_dim0
  g1[4] = (int)(dim1 | (dim2 << 16));           // tile_dim1 | tile_dim2
  g1[5] = (int)(unsigned)stride0;               // tensor_dim0_stride[31:0]
  g1[6] = (int)((unsigned)(stride0 >> 32) |
                ((unsigned)(stride1 & 0xFFFFull) << 16));
  g1[7] = (int)(unsigned)((stride1 >> 16) & 0xFFFFFFFFull);
  i32x4 g2;
  g2[0] = (int)dim2;                            // tensor_dim2
  g2[1] = 0; g2[2] = 0; g2[3] = 0;              // no dim3, no iterate
  i32x4 g3;
  g3[0] = 0; g3[1] = 0; g3[2] = 0; g3[3] = 0;
  i32x8 g4;
  g4[0] = 0; g4[1] = 0; g4[2] = 0; g4[3] = 0;
  g4[4] = 0; g4[5] = 0; g4[6] = 0; g4[7] = 0;
  __builtin_amdgcn_tensor_load_to_lds(g0, g1, g2, g3, g4, 0);
}

// ---------------------------------------------------------------------------
__global__ void lstm_init(unsigned* counter) {
  if (threadIdx.x == 0) *counter = 0u;
}

// ---------------------------------------------------------------------------
// Phase 1: xg = x @ W_ih^T, stored bf16. Block tile 64(M) x 256(N), K=512.
// Grid (2048, 8); 8 waves, wave w owns nt={2w,2w+1}, loops mt=0..3.
// ---------------------------------------------------------------------------
__global__ void __launch_bounds__(256)
xgates_gemm(const float* __restrict__ X, const float* __restrict__ Wih,
            bf16* __restrict__ XG) {
  __shared__ bf16 Xt[64 * 32];      // 4 KB
  __shared__ bf16 Wt[256 * 32];     // 16 KB
  __shared__ bf16 St[64 * 256];     // 32 KB output staging

  const int tid  = threadIdx.x;
  const int wv   = tid >> 5;
  const int lane = tid & 31;
  const int lr   = lane & 15;
  const int hi   = lane >> 4;
  const size_t mBase = (size_t)blockIdx.x * 64;
  const int    nBase = blockIdx.y * 256;

  v8f acc[4][2];
  for (int mt = 0; mt < 4; ++mt)
    for (int j = 0; j < 2; ++j)
      for (int v = 0; v < 8; ++v) acc[mt][j][v] = 0.0f;

  for (int kc = 0; kc < DIM / 32; ++kc) {
    // stage X tile: float4 -> 4 packed bf16 (b64 LDS store)
    for (int i = tid; i < 64 * 8; i += 256) {
      int r = i >> 3, kq = (i & 7) * 4;
      float4 v = *(const float4*)&X[(mBase + r) * DIM + kc * 32 + kq];
      uint2 p; p.x = pack_bf16x2(v.x, v.y); p.y = pack_bf16x2(v.z, v.w);
      *(uint2*)&Xt[r * 32 + kq] = p;
    }
    // stage W tile
    for (int i = tid; i < 256 * 8; i += 256) {
      int n = i >> 3, kq = (i & 7) * 4;
      float4 v = *(const float4*)&Wih[(size_t)(nBase + n) * DIM + kc * 32 + kq];
      uint2 p; p.x = pack_bf16x2(v.x, v.y); p.y = pack_bf16x2(v.z, v.w);
      *(uint2*)&Wt[n * 32 + kq] = p;
    }
    __syncthreads();

    const int nt0 = 2 * wv;
    for (int mt = 0; mt < 4; ++mt) {
      v16bf a  = lds_a_frag(Xt + (mt * 16 + lr) * 32 + hi * 16);
      v16bf b0 = lds_b_frag(Wt + ((nt0 + 0) * 16 + lr) * 32 + hi * 16);
      v16bf b1 = lds_b_frag(Wt + ((nt0 + 1) * 16 + lr) * 32 + hi * 16);
      acc[mt][0] = wmma_bf16(a, b0, acc[mt][0]);
      acc[mt][1] = wmma_bf16(a, b1, acc[mt][1]);
    }
    __syncthreads();
  }

  // stage results in LDS (D layout: M = v + 8*hi, N = lane&15) ...
  for (int mt = 0; mt < 4; ++mt)
    for (int j = 0; j < 2; ++j) {
      int nt = 2 * wv + j;
      for (int v = 0; v < 8; ++v)
        St[(mt * 16 + v + hi * 8) * 256 + nt * 16 + lr] =
            (bf16)acc[mt][j][v];
    }
  __syncthreads();
  // ... then coalesced b128 stores to global
  for (int i = tid; i < (64 * 256) / 8; i += 256) {
    int r = i >> 5, cq = (i & 31) * 8;
    *(uint4*)&XG[(mBase + r) * GATES + nBase + cq] =
        *(const uint4*)&St[r * 256 + cq];
  }
}

// ---------------------------------------------------------------------------
// Global monotonic barrier: counter reaches phase*NBLK when all blocks arrive.
// ---------------------------------------------------------------------------
static __device__ inline void grid_barrier(unsigned* c, unsigned phase) {
  __threadfence();
  __syncthreads();
  if (threadIdx.x == 0) {
    __hip_atomic_fetch_add(c, 1u, __ATOMIC_ACQ_REL, __HIP_MEMORY_SCOPE_AGENT);
    while (__hip_atomic_load(c, __ATOMIC_ACQUIRE, __HIP_MEMORY_SCOPE_AGENT) <
           phase * NBLK) {
      __builtin_amdgcn_s_sleep(2);
    }
  }
  __syncthreads();
}

// ---------------------------------------------------------------------------
// Phase 2: persistent recurrent scan. Block `blk` owns H-cols
// [blk*16, blk*16+16) -> gate cols g(n) = (n/16)*512 + blk*16 + (n%16).
// Per step: TDM stages h [64x512] and the xg slice [4][64][16] into LDS,
// WMMA computes gates = bias + xg + h @ Wh^T, fused f32 cell update.
// ---------------------------------------------------------------------------
__global__ void __launch_bounds__(256)
lstm_scan(const bf16*  __restrict__ XG,
          const float* __restrict__ Whh,
          const float* __restrict__ bih,
          const float* __restrict__ bhh,
          const float* __restrict__ hx,
          const float* __restrict__ cx,
          float*       __restrict__ out,
          bf16*        __restrict__ hbuf,   // 2 x [64 x 512] bf16
          unsigned*    __restrict__ counter) {
  __shared__ bf16  Wh[NCOL * HID];        // 64 KB, resident all kernel
  __shared__ bf16  Hs[BATCH * HID];       // 64 KB, TDM-staged per step
  __shared__ bf16  XGc[4 * BATCH * HC];   // 8 KB  [chunk][m][16], TDM-staged
  __shared__ float Gm[BATCH * NCOL];      // 16 KB gate accumulators
  __shared__ float Cs[BATCH * HC];        // 4 KB cell state slice (f32)
  __shared__ float Bia[NCOL];

  const int blk  = blockIdx.x;
  const int tid  = threadIdx.x;
  const int wv   = tid >> 5;
  const int lane = tid & 31;
  const int lr   = lane & 15;
  const int hi   = lane >> 4;

  // --- one-time setup: W_hh slice -> LDS bf16 (vectorized), bias, c0, h0 ---
  for (int i = tid; i < (NCOL * HID) / 4; i += 256) {
    int n = i >> 7, kq = (i & 127) * 4;
    int g = (n >> 4) * HID + blk * HC + (n & 15);
    float4 v = *(const float4*)&Whh[(size_t)g * HID + kq];
    uint2 p; p.x = pack_bf16x2(v.x, v.y); p.y = pack_bf16x2(v.z, v.w);
    *(uint2*)&Wh[n * HID + kq] = p;
  }
  if (tid < NCOL) {
    int g = (tid >> 4) * HID + blk * HC + (tid & 15);
    Bia[tid] = bih[g] + bhh[g];
  }
  for (int i = tid * 2; i < BATCH * HC; i += 512) {
    int m = i >> 4, jj = i & 15;
    Cs[i]     = cx[m * HID + blk * HC + jj];
    Cs[i + 1] = cx[m * HID + blk * HC + jj + 1];
    *(unsigned*)&hbuf[m * HID + blk * HC + jj] =
        pack_bf16x2(hx[m * HID + blk * HC + jj],
                    hx[m * HID + blk * HC + jj + 1]);
  }
  grid_barrier(counter, 1);   // all h0 slices visible everywhere

  const int mt  = wv >> 1;
  const int ntA = (wv & 1) * 2;
  const int ntB = ntA + 1;
  const unsigned lds_hs  = (unsigned)(unsigned long long)(const void*)Hs;
  const unsigned lds_xgc = (unsigned)(unsigned long long)(const void*)XGc;

  for (int t = 0; t < S_LEN; ++t) {
    const bf16* hcur = hbuf + (size_t)(t & 1) * BATCH * HID;
    bf16*       hnxt = hbuf + (size_t)((t + 1) & 1) * BATCH * HID;

    // --- TDM staging (wave 0 issues; TENSORcnt; publish via barrier) -----
    if (wv == 0) {
      // h: 2D tile, 512 contiguous bf16 x 64 rows, row stride 512
      tdm_load_bf16(lds_hs, (unsigned long long)(const void*)hcur,
                    512u, 64u, 0u, 512ull, 0ull);
      // xg slice: 3D tile, 16 bf16 x 64 rows (stride 2048) x 4 gate chunks
      // (stride 512) -> LDS [ch][m][16]
      tdm_load_bf16(lds_xgc,
                    (unsigned long long)(const void*)(XG +
                        (size_t)t * BATCH * GATES + (size_t)blk * HC),
                    16u, 64u, 4u, (unsigned long long)GATES, 512ull);
      __builtin_amdgcn_s_wait_tensorcnt(0);
    }
    __syncthreads();

    // --- WMMA: gates = bias + xg + h @ Wh^T ------------------------------
    v8f acc0, acc1;
    for (int v = 0; v < 8; ++v) {
      int m = mt * 16 + v + hi * 8;
      acc0[v] = Bia[ntA * 16 + lr] + (float)XGc[(ntA * BATCH + m) * HC + lr];
      acc1[v] = Bia[ntB * 16 + lr] + (float)XGc[(ntB * BATCH + m) * HC + lr];
    }
    for (int kc = 0; kc < HID / 32; ++kc) {
      v16bf a  = lds_a_frag(Hs + (mt * 16 + lr) * HID + kc * 32 + hi * 16);
      v16bf b0 = lds_b_frag(Wh + (ntA * 16 + lr) * HID + kc * 32 + hi * 16);
      v16bf b1 = lds_b_frag(Wh + (ntB * 16 + lr) * HID + kc * 32 + hi * 16);
      acc0 = wmma_bf16(a, b0, acc0);
      acc1 = wmma_bf16(a, b1, acc1);
    }
    for (int v = 0; v < 8; ++v) {
      int m = mt * 16 + v + hi * 8;
      Gm[m * NCOL + ntA * 16 + lr] = acc0[v];
      Gm[m * NCOL + ntB * 16 + lr] = acc1[v];
    }
    __syncthreads();

    // --- fused elementwise, 2 columns per iteration (paired stores) ------
    for (int i = tid * 2; i < BATCH * HC; i += 512) {
      int m = i >> 4, jj = i & 15;
      const float* gr = &Gm[m * NCOL + jj];
      float gi0 = sigmoidf_(gr[0]),  gi1 = sigmoidf_(gr[1]);
      float gf0 = sigmoidf_(gr[16]), gf1 = sigmoidf_(gr[17]);
      float gg0 = tanhf(gr[32]),     gg1 = tanhf(gr[33]);
      float go0 = sigmoidf_(gr[48]), go1 = sigmoidf_(gr[49]);
      float c0 = gf0 * Cs[i] + gi0 * gg0;
      float c1 = gf1 * Cs[i + 1] + gi1 * gg1;
      float h0 = go0 * tanhf(c0);
      float h1 = go1 * tanhf(c1);
      Cs[i] = c0; Cs[i + 1] = c1;
      int    j = blk * HC + jj;
      size_t o = (size_t)t * BATCH * HID + (size_t)m * HID + j;
      float2 hv; hv.x = h0; hv.y = h1;
      float2 cv; cv.x = c0; cv.y = c1;
      *(float2*)&out[o] = hv;                                    // hs
      *(float2*)&out[(size_t)S_LEN * BATCH * HID + o] = cv;      // cs
      *(unsigned*)&hnxt[m * HID + j] = pack_bf16x2(h0, h1);
    }
    grid_barrier(counter, (unsigned)(t + 2));
  }
}

// ---------------------------------------------------------------------------
extern "C" void kernel_launch(void* const* d_in, const int* in_sizes, int n_in,
                              void* d_out, int out_size, void* d_ws,
                              size_t ws_size, hipStream_t stream) {
  const float* x   = (const float*)d_in[0];  // [S,B,D]
  const float* hx  = (const float*)d_in[1];  // [B,H]
  const float* cx  = (const float*)d_in[2];  // [B,H]
  const float* Wih = (const float*)d_in[3];  // [4H,D]
  const float* Whh = (const float*)d_in[4];  // [4H,H]
  const float* bih = (const float*)d_in[5];  // [4H]
  const float* bhh = (const float*)d_in[6];  // [4H]
  float* out = (float*)d_out;                // hs[S,B,H] ++ cs[S,B,H]

  const size_t XG_BYTES   = (size_t)S_LEN * BATCH * GATES * sizeof(bf16);
  const size_t HBUF_BYTES = (size_t)2 * BATCH * HID * sizeof(bf16);

  char*     ws      = (char*)d_ws;
  bf16*     XG      = (bf16*)ws;
  bf16*     hbuf    = (bf16*)(ws + XG_BYTES);
  unsigned* counter = (unsigned*)(ws + XG_BYTES + HBUF_BYTES);

  (void)in_sizes; (void)n_in; (void)out_size; (void)ws_size;

  lstm_init<<<1, 32, 0, stream>>>(counter);
  xgates_gemm<<<dim3((S_LEN * BATCH) / 64, GATES / 256), 256, 0, stream>>>(
      x, Wih, XG);
  lstm_scan<<<NBLK, 256, 0, stream>>>(XG, Whh, bih, bhh, hx, cx, out, hbuf,
                                      counter);
}
